// KalmanNetNN_70901320123081
// MI455X (gfx1250) — matrix operand order repaired and checked by reading, not compiled, not run on hardware
//
#include <hip/hip_runtime.h>
#include <math.h>

typedef __bf16 bf16;
typedef __attribute__((ext_vector_type(16))) __bf16 v16bf;
typedef __attribute__((ext_vector_type(8)))  float  v8f;
typedef __attribute__((ext_vector_type(4)))  unsigned int v4u;

#define BSZ 4096

// ---------------- CDNA5 async global->LDS copy (ASYNCcnt) --------------------
__device__ __forceinline__ void async_b128(unsigned lds_addr, const void* g) {
  // GLOBAL_LOAD_ASYNC_TO_LDS_B128: vdst = LDS byte address, vaddr = 64-bit global
  asm volatile("global_load_async_to_lds_b128 %0, %1, off"
               :: "v"(lds_addr), "v"(g) : "memory");
}
__device__ __forceinline__ void wait_async0() {
  asm volatile("s_wait_asynccnt 0x0" ::: "memory");
}

// ---------------- generic bf16-WMMA GEMM: out = act(A @ W^T + bias) ----------
// A: [M][lda] bf16, W: [Npad][ldw] bf16 (pre-converted, rows zero-padded to 64).
// M multiple of 256, K multiple of 32. Block tile 256x64, 8 waves of 64x32.
#define BM 256
#define BN 64
#define BK 32
#define RS 80                       // LDS row pitch bytes (64B data + 16B pad)
#define A_LDS_BYTES (BM * RS)       // 20480
#define W_LDS_BYTES (BN * RS)       // 5120
#define STAGE_BYTES (A_LDS_BYTES + W_LDS_BYTES)

union Frag { v16bf bf; v4u u[2]; };

__global__ __launch_bounds__(256)
void gemm_bf16_wmma(const bf16* __restrict__ A, int lda,
                    const bf16* __restrict__ Wb, int ldw,
                    const float* __restrict__ bias,
                    int N, int K, int relu,
                    float* __restrict__ outF, int ldf,
                    bf16* __restrict__ outB, int ldb)
{
  __shared__ __align__(16) unsigned char lds[2][STAGE_BYTES];
  const int t   = threadIdx.x;
  const int bm0 = blockIdx.x * BM;
  const int bn0 = blockIdx.y * BN;

  unsigned sbase[2];
  sbase[0] = (unsigned)(size_t)(void*)&lds[0][0];
  sbase[1] = (unsigned)(size_t)(void*)&lds[1][0];

  // stage A (256x32 bf16) + W (64x32 bf16) tiles via async copies
  auto stage = [&](int buf, int kk) {
#pragma unroll
    for (int i2 = 0; i2 < 4; ++i2) {
      int c = t + 256 * i2;
      int r = c >> 2, q = c & 3;
      async_b128(sbase[buf] + r * RS + q * 16,
                 A + (size_t)(bm0 + r) * lda + kk + q * 8);
    }
    {
      int r = t >> 2, q = t & 3;
      async_b128(sbase[buf] + A_LDS_BYTES + r * RS + q * 16,
                 Wb + (size_t)(bn0 + r) * ldw + kk + q * 8);
    }
  };

  const int lane = t & 31;
  const int wv = t >> 5;
  const int wM = wv & 3, wN = wv >> 2;     // 4 waves along M, 2 along N
  const int lr = lane & 15, lh = lane >> 4;

  v8f acc[4][2];
  for (int i = 0; i < 4; ++i)
    for (int j = 0; j < 2; ++j)
      for (int v = 0; v < 8; ++v) acc[i][j][v] = 0.f;

  const int KT = K / BK;
  int buf = 0;
  stage(0, 0);
  wait_async0();
  __syncthreads();

  for (int kt = 0; kt < KT; ++kt) {
    if (kt + 1 < KT) stage(buf ^ 1, (kt + 1) * BK);

    const unsigned char* sA = lds[buf];
    const unsigned char* sW = lds[buf] + A_LDS_BYTES;
    Frag afr[4], bfr[2];
#pragma unroll
    for (int i = 0; i < 4; ++i) {
      // A 16x32 bf16 fragment: lane = M row, lane-half selects K interleave
      const unsigned char* pa = sA + (wM * 64 + i * 16 + lr) * RS + lh * 16;
      afr[i].u[0] = *(const v4u*)pa;        // VGPR0-3: K = 8*lh .. 8*lh+7
      afr[i].u[1] = *(const v4u*)(pa + 32); // VGPR4-7: K = 16+8*lh ..
    }
#pragma unroll
    for (int j = 0; j < 2; ++j) {
      // B 32x16 bf16 fragment: lane = N col, lane-half selects K interleave (same as A)
      const unsigned char* pb = sW + (wN * 32 + j * 16 + lr) * RS + lh * 16;
      bfr[j].u[0] = *(const v4u*)pb;
      bfr[j].u[1] = *(const v4u*)(pb + 32);
    }
#pragma unroll
    for (int i = 0; i < 4; ++i)
#pragma unroll
      for (int j = 0; j < 2; ++j)
        acc[i][j] = __builtin_amdgcn_wmma_f32_16x16x32_bf16(
            false, afr[i].bf, false, bfr[j].bf, (short)0, acc[i][j], false, false);

    wait_async0();        // next-stage copies complete (overlapped with WMMAs)
    __syncthreads();
    buf ^= 1;
  }

  // epilogue: C layout VGPR v -> M = v + 8*laneHi, N = lane%16
#pragma unroll
  for (int i = 0; i < 4; ++i) {
#pragma unroll
    for (int j = 0; j < 2; ++j) {
      int col = bn0 + wN * 32 + j * 16 + lr;
      if (col < N) {
        float bv = bias ? bias[col] : 0.f;
        int mbase = bm0 + wM * 64 + i * 16 + lh * 8;
#pragma unroll
        for (int v = 0; v < 8; ++v) {
          float x = acc[i][j][v] + bv;
          if (relu) x = fmaxf(x, 0.f);
          long row = mbase + v;
          if (outF) outF[row * (long)ldf + col] = x;
          if (outB) outB[row * (long)ldb + col] = (bf16)x;
        }
      }
    }
  }
}

// ---------------- conversion kernels -----------------------------------------
__global__ void f32_to_bf16_k(const float* __restrict__ s, bf16* __restrict__ d, int n) {
  int i = blockIdx.x * blockDim.x + threadIdx.x;
  if (i < n) d[i] = (bf16)s[i];
}

// weight convert with row zero-padding to rowsPad
__global__ void conv_w_k(const float* __restrict__ s, bf16* __restrict__ d,
                         int rows, int cols, int rowsPad) {
  long i = blockIdx.x * (long)blockDim.x + threadIdx.x;
  long tot = (long)rowsPad * cols;
  if (i >= tot) return;
  long r = i / cols;
  d[i] = (r < rows) ? (bf16)s[i] : (bf16)0.f;
}

// ---------------- elementwise kernels (f32 math) -----------------------------
__device__ inline void l2norm_store(bf16* dst, const float* v, int n) {
  float ss = 0.f;
  for (int k = 0; k < n; ++k) ss += v[k] * v[k];
  float nm = sqrtf(ss);
  float inv = 1.f / fmaxf(nm, 1e-12f);
  for (int k = 0; k < n; ++k) dst[k] = (bf16)(v[k] * inv);
}

__global__ void prep_k(const float* __restrict__ y, const float* __restrict__ xpost,
                       const float* __restrict__ xpost_prev, const float* __restrict__ xprior_prev,
                       const float* __restrict__ yprev,
                       const float* __restrict__ F, const float* __restrict__ Hm,
                       float* __restrict__ m1x_prior, float* __restrict__ m1y,
                       bf16* __restrict__ a_upd, bf16* __restrict__ a_evol,
                       bf16* __restrict__ a_obs, int Btot)
{
  int b = blockIdx.x * blockDim.x + threadIdx.x;
  if (b >= Btot) return;
  float xp[32], pr[32], my[16], d[32];
  for (int k = 0; k < 32; ++k) xp[k] = xpost[b * 32 + k];
  for (int i = 0; i < 32; ++i) {
    float s = 0.f;
    for (int j = 0; j < 32; ++j) s += F[i * 32 + j] * xp[j];
    pr[i] = s; m1x_prior[b * 32 + i] = s;
  }
  for (int i = 0; i < 16; ++i) {
    float s = 0.f;
    for (int j = 0; j < 32; ++j) s += Hm[i * 32 + j] * pr[j];
    my[i] = s; m1y[b * 16 + i] = s;
  }
  for (int k = 0; k < 32; ++k) d[k] = xp[k] - xprior_prev[b * 32 + k];
  l2norm_store(a_upd + b * 32, d, 32);
  for (int k = 0; k < 32; ++k) d[k] = xp[k] - xpost_prev[b * 32 + k];
  l2norm_store(a_evol + b * 32, d, 32);
  for (int k = 0; k < 16; ++k) d[k] = y[b * 16 + k] - yprev[b * 16 + k];
  l2norm_store(a_obs + b * 32, d, 16);
  for (int k = 0; k < 16; ++k) d[k] = y[b * 16 + k] - my[k];
  l2norm_store(a_obs + b * 32 + 16, d, 16);
}

__device__ inline float sigm(float x) { return 1.f / (1.f + expf(-x)); }

__global__ void gru_gate_k(const float* __restrict__ gi, const float* __restrict__ gh,
                           const float* __restrict__ hprev, int H, int Btot,
                           float* __restrict__ outF,
                           bf16* __restrict__ b0, int ld0,
                           bf16* __restrict__ b1, int ld1,
                           bf16* __restrict__ b2, int ld2)
{
  int i = blockIdx.x * blockDim.x + threadIdx.x;
  if (i >= Btot * H) return;
  int b = i / H, j = i - b * H;
  long g = (long)b * 3 * H;
  float r = sigm(gi[g + j]         + gh[g + j]);
  float z = sigm(gi[g + H + j]     + gh[g + H + j]);
  float c = tanhf(gi[g + 2 * H + j] + r * gh[g + 2 * H + j]);
  float h = hprev[(long)b * H + j];
  float o = (1.f - z) * c + z * h;
  if (outF) outF[(long)b * H + j] = o;
  bf16 ob = (bf16)o;
  if (b0) b0[(long)b * ld0 + j] = ob;
  if (b1) b1[(long)b * ld1 + j] = ob;
  if (b2) b2[(long)b * ld2 + j] = ob;
}

__global__ void final_k(const float* __restrict__ kg, const float* __restrict__ y,
                        const float* __restrict__ m1y, const float* __restrict__ m1x_prior,
                        float* __restrict__ out, int Btot)
{
  int i = blockIdx.x * blockDim.x + threadIdx.x;
  if (i >= Btot * 32) return;
  int b = i >> 5, m = i & 31;
  float acc = m1x_prior[b * 32 + m];
  const float* K = kg + (long)b * 512 + m * 16;
  for (int n = 0; n < 16; ++n) acc += K[n] * (y[b * 16 + n] - m1y[b * 16 + n]);
  out[b * 32 + m] = acc;
}

// ---------------- host orchestration ----------------------------------------
extern "C" void kernel_launch(void* const* d_in, const int* in_sizes, int n_in,
                              void* d_out, int out_size, void* d_ws, size_t ws_size,
                              hipStream_t stream) {
  (void)in_sizes; (void)n_in; (void)out_size; (void)ws_size;
  const float* y      = (const float*)d_in[0];
  const float* xpost  = (const float*)d_in[1];
  const float* xpostp = (const float*)d_in[2];
  const float* xpriop = (const float*)d_in[3];
  const float* yprev  = (const float*)d_in[4];
  const float* hQ     = (const float*)d_in[5];
  const float* hSig   = (const float*)d_in[6];
  const float* hS     = (const float*)d_in[7];
  const float* F_mat  = (const float*)d_in[8];
  const float* H_mat  = (const float*)d_in[9];
  const float* Qwih = (const float*)d_in[10], *Qwhh = (const float*)d_in[11];
  const float* Qbih = (const float*)d_in[12], *Qbhh = (const float*)d_in[13];
  const float* Gwih = (const float*)d_in[14], *Gwhh = (const float*)d_in[15];
  const float* Gbih = (const float*)d_in[16], *Gbhh = (const float*)d_in[17];
  const float* Swih = (const float*)d_in[18], *Swhh = (const float*)d_in[19];
  const float* Sbih = (const float*)d_in[20], *Sbhh = (const float*)d_in[21];
  const float* fc1w = (const float*)d_in[22], *fc1b = (const float*)d_in[23];
  const float* fc2aw= (const float*)d_in[24], *fc2ab= (const float*)d_in[25];
  const float* fc2bw= (const float*)d_in[26], *fc2bb= (const float*)d_in[27];
  const float* fc3w = (const float*)d_in[28], *fc3b = (const float*)d_in[29];
  const float* fc4w = (const float*)d_in[30], *fc4b = (const float*)d_in[31];
  const float* fc5w = (const float*)d_in[32], *fc5b = (const float*)d_in[33];
  const float* fc6w = (const float*)d_in[34], *fc6b = (const float*)d_in[35];
  const float* fc7w = (const float*)d_in[36], *fc7b = (const float*)d_in[37];

  // workspace carve (256B aligned)
  size_t off = 0;
  auto alloc = [&](size_t bytes) { size_t o = off; off = (off + bytes + 255) & ~(size_t)255; return o; };
  auto padN  = [](int n) { return (n + 63) & ~63; };
  char* base = (char*)d_ws;
  // f32 scratch
  size_t o_prior = alloc((size_t)BSZ * 32 * 4);
  size_t o_m1y   = alloc((size_t)BSZ * 16 * 4);
  size_t o_G0    = alloc((size_t)BSZ * 3072 * 4);
  size_t o_G1    = alloc((size_t)BSZ * 3072 * 4);
  size_t o_fc2   = alloc((size_t)BSZ * 512 * 4);
  // bf16 activations
  size_t o_upd   = alloc((size_t)BSZ * 32 * 2);
  size_t o_evol  = alloc((size_t)BSZ * 32 * 2);
  size_t o_obs   = alloc((size_t)BSZ * 32 * 2);
  size_t o_fc5   = alloc((size_t)BSZ * 160 * 2);
  size_t o_hQ    = alloc((size_t)BSZ * 1024 * 2);
  size_t o_hSig  = alloc((size_t)BSZ * 1024 * 2);
  size_t o_hS    = alloc((size_t)BSZ * 256 * 2);
  size_t o_insig = alloc((size_t)BSZ * 1184 * 2);
  size_t o_osig  = alloc((size_t)BSZ * 1024 * 2);
  size_t o_infc2 = alloc((size_t)BSZ * 1280 * 2);
  size_t o_infc4 = alloc((size_t)BSZ * 2048 * 2);
  size_t o_ins   = alloc((size_t)BSZ * 416 * 2);
  size_t o_mid   = alloc((size_t)BSZ * 2560 * 2);
  size_t o_infc3 = alloc((size_t)BSZ * 768 * 2);
  // bf16 weights (rows padded to 64)
  size_t o_wQi = alloc((size_t)padN(3072) * 160 * 2);
  size_t o_wQh = alloc((size_t)padN(3072) * 1024 * 2);
  size_t o_wGi = alloc((size_t)padN(3072) * 1184 * 2);
  size_t o_wGh = alloc((size_t)padN(3072) * 1024 * 2);
  size_t o_wSi = alloc((size_t)padN(768) * 416 * 2);
  size_t o_wSh = alloc((size_t)padN(768) * 256 * 2);
  size_t o_w1  = alloc((size_t)padN(256) * 1024 * 2);
  size_t o_w2a = alloc((size_t)padN(2560) * 1280 * 2);
  size_t o_w2b = alloc((size_t)padN(512) * 2560 * 2);
  size_t o_w3  = alloc((size_t)padN(1024) * 768 * 2);
  size_t o_w4  = alloc((size_t)padN(1024) * 2048 * 2);
  size_t o_w5  = alloc((size_t)padN(160) * 32 * 2);
  size_t o_w6  = alloc((size_t)padN(160) * 32 * 2);
  size_t o_w7  = alloc((size_t)padN(160) * 32 * 2);

  float* m1x_prior = (float*)(base + o_prior);
  float* m1y       = (float*)(base + o_m1y);
  float* G0        = (float*)(base + o_G0);
  float* G1        = (float*)(base + o_G1);
  float* fc2out    = (float*)(base + o_fc2);
  bf16* a_upd  = (bf16*)(base + o_upd);
  bf16* a_evol = (bf16*)(base + o_evol);
  bf16* a_obs  = (bf16*)(base + o_obs);
  bf16* a_fc5  = (bf16*)(base + o_fc5);
  bf16* a_hQ   = (bf16*)(base + o_hQ);
  bf16* a_hSig = (bf16*)(base + o_hSig);
  bf16* a_hS   = (bf16*)(base + o_hS);
  bf16* a_insig= (bf16*)(base + o_insig);
  bf16* a_osig = (bf16*)(base + o_osig);
  bf16* a_infc2= (bf16*)(base + o_infc2);
  bf16* a_infc4= (bf16*)(base + o_infc4);
  bf16* a_ins  = (bf16*)(base + o_ins);
  bf16* a_mid  = (bf16*)(base + o_mid);
  bf16* a_infc3= (bf16*)(base + o_infc3);
  bf16* wQi = (bf16*)(base + o_wQi); bf16* wQh = (bf16*)(base + o_wQh);
  bf16* wGi = (bf16*)(base + o_wGi); bf16* wGh = (bf16*)(base + o_wGh);
  bf16* wSi = (bf16*)(base + o_wSi); bf16* wSh = (bf16*)(base + o_wSh);
  bf16* w1  = (bf16*)(base + o_w1);  bf16* w2a = (bf16*)(base + o_w2a);
  bf16* w2b = (bf16*)(base + o_w2b); bf16* w3  = (bf16*)(base + o_w3);
  bf16* w4  = (bf16*)(base + o_w4);  bf16* w5  = (bf16*)(base + o_w5);
  bf16* w6  = (bf16*)(base + o_w6);  bf16* w7  = (bf16*)(base + o_w7);

  float* out_x    = (float*)d_out;
  float* out_hQ   = out_x + (size_t)BSZ * 32;
  float* out_hSig = out_hQ + (size_t)BSZ * 1024;
  float* out_hS   = out_hSig + (size_t)BSZ * 1024;

  auto gemm = [&](const bf16* A, int lda, const bf16* Wb, const float* bias,
                  int N, int K, int relu, float* oF, int ldf, bf16* oB, int ldb) {
    dim3 g(BSZ / BM, (N + BN - 1) / BN);
    gemm_bf16_wmma<<<g, 256, 0, stream>>>(A, lda, Wb, K, bias, N, K, relu, oF, ldf, oB, ldb);
  };
  auto conv = [&](const float* s, bf16* d, int n) {
    f32_to_bf16_k<<<(n + 255) / 256, 256, 0, stream>>>(s, d, n);
  };
  auto convW = [&](const float* s, bf16* d, int rows, int cols) {
    long tot = (long)padN(rows) * cols;
    conv_w_k<<<(int)((tot + 255) / 256), 256, 0, stream>>>(s, d, rows, cols, padN(rows));
  };

  // 0) weight pre-conversion f32 -> bf16 (padded rows)
  convW(Qwih, wQi, 3072, 160);   convW(Qwhh, wQh, 3072, 1024);
  convW(Gwih, wGi, 3072, 1184);  convW(Gwhh, wGh, 3072, 1024);
  convW(Swih, wSi, 768, 416);    convW(Swhh, wSh, 768, 256);
  convW(fc1w, w1, 256, 1024);    convW(fc2aw, w2a, 2560, 1280);
  convW(fc2bw, w2b, 512, 2560);  convW(fc3w, w3, 1024, 768);
  convW(fc4w, w4, 1024, 2048);   convW(fc5w, w5, 160, 32);
  convW(fc6w, w6, 160, 32);      convW(fc7w, w7, 160, 32);

  // 1) f32 -> bf16 copies of GRU hidden states
  conv(hQ,   a_hQ,   BSZ * 1024);
  conv(hSig, a_hSig, BSZ * 1024);
  conv(hS,   a_hS,   BSZ * 256);
  // 2) prior/innovation features
  prep_k<<<(BSZ + 255) / 256, 256, 0, stream>>>(y, xpost, xpostp, xpriop, yprev,
                                                F_mat, H_mat, m1x_prior, m1y,
                                                a_upd, a_evol, a_obs, BSZ);
  // 3) fc5 + GRU-Q
  gemm(a_upd, 32, w5, fc5b, 160, 32, 1, nullptr, 0, a_fc5, 160);
  gemm(a_fc5, 160,  wQi, Qbih, 3072, 160,  0, G0, 3072, nullptr, 0);
  gemm(a_hQ,  1024, wQh, Qbhh, 3072, 1024, 0, G1, 3072, nullptr, 0);
  gru_gate_k<<<(BSZ * 1024 + 255) / 256, 256, 0, stream>>>(G0, G1, hQ, 1024, BSZ,
      out_hQ, a_insig, 1184, nullptr, 0, nullptr, 0);
  // 4) fc6 -> in_sigma tail, GRU-Sigma
  gemm(a_evol, 32, w6, fc6b, 160, 32, 1, nullptr, 0, a_insig + 1024, 1184);
  gemm(a_insig, 1184, wGi, Gbih, 3072, 1184, 0, G0, 3072, nullptr, 0);
  gemm(a_hSig,  1024, wGh, Gbhh, 3072, 1024, 0, G1, 3072, nullptr, 0);
  gru_gate_k<<<(BSZ * 1024 + 255) / 256, 256, 0, stream>>>(G0, G1, hSig, 1024, BSZ,
      nullptr, a_osig, 1024, a_infc2, 1280, a_infc4, 2048);
  // 5) fc1 + fc7 -> in_s, GRU-S
  gemm(a_osig, 1024, w1, fc1b, 256, 1024, 1, nullptr, 0, a_ins, 416);
  gemm(a_obs,  32,   w7, fc7b, 160, 32,   1, nullptr, 0, a_ins + 256, 416);
  gemm(a_ins, 416, wSi, Sbih, 768, 416, 0, G0, 768, nullptr, 0);
  gemm(a_hS,  256, wSh, Sbhh, 768, 256, 0, G1, 768, nullptr, 0);
  gru_gate_k<<<(BSZ * 256 + 255) / 256, 256, 0, stream>>>(G0, G1, hS, 256, BSZ,
      out_hS, a_infc2 + 1024, 1280, a_infc3, 768, nullptr, 0);
  // 6) fc2a -> fc2b (KGain), fc3, fc4 (h_Sigma_new)
  gemm(a_infc2, 1280, w2a, fc2ab, 2560, 1280, 1, nullptr, 0, a_mid, 2560);
  gemm(a_mid,   2560, w2b, fc2bb, 512,  2560, 0, fc2out, 512, a_infc3 + 256, 768);
  gemm(a_infc3, 768,  w3,  fc3b,  1024, 768,  1, nullptr, 0, a_infc4 + 1024, 2048);
  gemm(a_infc4, 2048, w4,  fc4b,  1024, 2048, 1, out_hSig, 1024, nullptr, 0);
  // 7) Kalman update
  final_k<<<(BSZ * 32 + 255) / 256, 256, 0, stream>>>(fc2out, y, m1y, m1x_prior, out_x, BSZ);
}
